// FFFFanout_66013647339602
// MI455X (gfx1250) — matrix operand order repaired and checked by reading, not compiled, not run on hardware
//
#include <hip/hip_runtime.h>
#include <math.h>

#define D     1024
#define P_    4
#define FAN   4
#define G_    1365
#define NPG   5460      /* P_*G_ == HID/FAN */
#define HID   21840
#define BATCH 4096

/* ------------------------------ common helpers ------------------------------ */

typedef unsigned short u16;
typedef __attribute__((ext_vector_type(4)))  u16    u16x4;
typedef __attribute__((ext_vector_type(8)))  u16    u16x8;
typedef __attribute__((ext_vector_type(16))) u16    u16x16;
typedef __attribute__((ext_vector_type(16))) __bf16 v16bf;
typedef __attribute__((ext_vector_type(8)))  float  v8f;

static __device__ __forceinline__ u16 f32_to_bf16_rne(float f) {
    unsigned u = __builtin_bit_cast(unsigned, f);
    unsigned r = u + 0x7FFFu + ((u >> 16) & 1u);
    return (u16)(r >> 16);
}
static __device__ __forceinline__ float bf16_to_f32(u16 h) {
    unsigned u = ((unsigned)h) << 16;
    return __builtin_bit_cast(float, u);
}
static __device__ __forceinline__ float gelu_exact(float x) {
    return 0.5f * x * (1.0f + erff(x * 0.70710678118654752f));
}
/* build a 16-element bf16 fragment from two contiguous 16B LDS chunks */
static __device__ __forceinline__ v16bf load_frag(const u16* p0, const u16* p1) {
    u16x8 a = *(const u16x8*)p0;
    u16x8 b = *(const u16x8*)p1;
    u16x16 c = __builtin_shufflevector(a, b, 0,1,2,3,4,5,6,7,8,9,10,11,12,13,14,15);
    return __builtin_bit_cast(v16bf, c);
}

/* ------------------------- kernel 0: transpose W_out ------------------------ */
/* W_out [D, HID] row-major -> Wt [HID, D] row-major (contiguous column gather) */
__global__ void transpose_wout(const float* __restrict__ Wout, float* __restrict__ Wt) {
    __shared__ float t[32][33];
    const int hx = blockIdx.x * 32 + threadIdx.x;   /* HID coord */
    const int dy = blockIdx.y * 32 + threadIdx.y;   /* D coord   */
#pragma unroll
    for (int i = 0; i < 32; i += 8) {
        if (hx < HID) t[threadIdx.y + i][threadIdx.x] = Wout[(size_t)(dy + i) * HID + hx];
    }
    __syncthreads();
    const int ht = blockIdx.x * 32 + threadIdx.y;
    const int dt = blockIdx.y * 32 + threadIdx.x;
#pragma unroll
    for (int i = 0; i < 32; i += 8) {
        const int h = ht + i;
        if (h < HID) Wt[(size_t)h * D + dt] = t[threadIdx.x][threadIdx.y + i];
    }
}

/* --------------- kernel 1: GEMM + gelu + per-group argmax (WMMA) ----------- */
/* C[4096, 21840] = x @ W_in^T, bf16 hi/lo split (3 WMMAs) for ~fp32 accuracy.
   Block tile 128(M) x 80(N), BK=32. 8 waves, wave = 16 rows x 5 N-tiles.
   Emits only decision bytes dec[b*5460 + h/4] = argmax over fan-of-4. */
#define BM  128
#define BN  80
#define BK  32
#define LDK 40   /* LDS row stride (u16) : 80B, 16B-aligned, bank-staggered */

__global__ void __launch_bounds__(256)
gemm_decide(const float* __restrict__ x, const float* __restrict__ W_in,
            const float* __restrict__ b_in, unsigned char* __restrict__ dec) {
    __shared__ __align__(16) u16 Ahi[BM * LDK];
    __shared__ __align__(16) u16 Alo[BM * LDK];
    __shared__ __align__(16) u16 Bhi[BN * LDK];
    __shared__ __align__(16) u16 Blo[BN * LDK];
    __shared__ float bs[BN];

    const int tid  = threadIdx.x;
    const int wave = tid >> 5;
    const int lane = tid & 31;
    const int n0 = blockIdx.x * BN;      /* 273 blocks: 273*80 = 21840 */
    const int m0 = blockIdx.y * BM;      /* 32 blocks : 32*128 = 4096  */

    if (tid < BN) bs[tid] = b_in[n0 + tid];

    v8f acc[5] = {};

    const int arow = wave * 16 + (lane & 15);
    const int akb  = (lane < 16) ? 0 : 8;
    const int brow = lane & 15;
    const int bkb  = (lane < 16) ? 0 : 8;

    for (int kk = 0; kk < D / BK; ++kk) {
        const int k0 = kk * BK;
        __syncthreads();
        /* stage A tile: 128 x 32 f32 -> hi/lo bf16 (1024 float4, 4 per thread) */
#pragma unroll
        for (int i = 0; i < 4; ++i) {
            const int idx = tid + i * 256;
            const int row = idx >> 3, c4 = idx & 7;
            const float4 f = *(const float4*)(x + (size_t)(m0 + row) * D + k0 + c4 * 4);
            u16 h0 = f32_to_bf16_rne(f.x), h1 = f32_to_bf16_rne(f.y);
            u16 h2 = f32_to_bf16_rne(f.z), h3 = f32_to_bf16_rne(f.w);
            u16 l0 = f32_to_bf16_rne(f.x - bf16_to_f32(h0));
            u16 l1 = f32_to_bf16_rne(f.y - bf16_to_f32(h1));
            u16 l2 = f32_to_bf16_rne(f.z - bf16_to_f32(h2));
            u16 l3 = f32_to_bf16_rne(f.w - bf16_to_f32(h3));
            *(u16x4*)&Ahi[row * LDK + c4 * 4] = (u16x4){h0, h1, h2, h3};
            *(u16x4*)&Alo[row * LDK + c4 * 4] = (u16x4){l0, l1, l2, l3};
        }
        /* stage B tile: 80 x 32 (640 float4) */
#pragma unroll
        for (int i = 0; i < 3; ++i) {
            const int idx = tid + i * 256;
            if (idx < (BN * BK) / 4) {
                const int row = idx >> 3, c4 = idx & 7;
                const float4 f = *(const float4*)(W_in + (size_t)(n0 + row) * D + k0 + c4 * 4);
                u16 h0 = f32_to_bf16_rne(f.x), h1 = f32_to_bf16_rne(f.y);
                u16 h2 = f32_to_bf16_rne(f.z), h3 = f32_to_bf16_rne(f.w);
                u16 l0 = f32_to_bf16_rne(f.x - bf16_to_f32(h0));
                u16 l1 = f32_to_bf16_rne(f.y - bf16_to_f32(h1));
                u16 l2 = f32_to_bf16_rne(f.z - bf16_to_f32(h2));
                u16 l3 = f32_to_bf16_rne(f.w - bf16_to_f32(h3));
                *(u16x4*)&Bhi[row * LDK + c4 * 4] = (u16x4){h0, h1, h2, h3};
                *(u16x4*)&Blo[row * LDK + c4 * 4] = (u16x4){l0, l1, l2, l3};
            }
        }
        __syncthreads();
        /* prefetch next K tile while this one computes */
        if (kk + 1 < D / BK) {
            const int row = tid >> 3, c4 = tid & 7;
            __builtin_prefetch(x + (size_t)(m0 + row) * D + k0 + BK + c4 * 4, 0, 0);
            if (row < BN)
                __builtin_prefetch(W_in + (size_t)(n0 + row) * D + k0 + BK + c4 * 4, 0, 0);
        }

        /* A fragment (ISA 16-bit A 16x32 layout: lanes 0-15 rows, K halves split at lane 16) */
        const v16bf a_hi = load_frag(&Ahi[arow * LDK + akb], &Ahi[arow * LDK + akb + 16]);
        const v16bf a_lo = load_frag(&Alo[arow * LDK + akb], &Alo[arow * LDK + akb + 16]);
#pragma unroll
        for (int nt = 0; nt < 5; ++nt) {
            const int r = (nt * 16 + brow) * LDK + bkb;
            const v16bf b_hi = load_frag(&Bhi[r], &Bhi[r + 16]);
            const v16bf b_lo = load_frag(&Blo[r], &Blo[r + 16]);
            acc[nt] = __builtin_amdgcn_wmma_f32_16x16x32_bf16(false, a_hi, false, b_hi,
                                                              (short)0, acc[nt], false, false);
            acc[nt] = __builtin_amdgcn_wmma_f32_16x16x32_bf16(false, a_hi, false, b_lo,
                                                              (short)0, acc[nt], false, false);
            acc[nt] = __builtin_amdgcn_wmma_f32_16x16x32_bf16(false, a_lo, false, b_hi,
                                                              (short)0, acc[nt], false, false);
        }
    }

    /* epilogue: bias + exact gelu + argmax over each fan-of-4 quad (cross-lane) */
#pragma unroll
    for (int nt = 0; nt < 5; ++nt) {
        const float bias = bs[nt * 16 + (lane & 15)];
#pragma unroll
        for (int v = 0; v < 8; ++v) {
            float g = gelu_exact(acc[nt][v] + bias);
            int idx = lane & 3;
            {   /* xor-1 partner */
                float g1 = __shfl_xor(g, 1);
                int   i1 = idx ^ 1;
                if (g1 > g || (g1 == g && i1 < idx)) { g = g1; idx = i1; }
            }
            {   /* xor-2 partner (carries its updated winner) */
                float g2 = __shfl_xor(g, 2);
                int   i2 = __shfl_xor(idx, 2);
                if (g2 > g || (g2 == g && i2 < idx)) { g = g2; idx = i2; }
            }
            if ((lane & 3) == 0) {
                const int b = m0 + wave * 16 + ((lane < 16) ? v : v + 8);
                const int h = n0 + nt * 16 + (lane & 15);
                dec[(size_t)b * NPG + (h >> 2)] = (unsigned char)idx;
            }
        }
    }
}

/* ------- kernel 2: tree walk + sparse activation recompute + output -------- */
/* one block per batch row b; 96 active hid columns per row */
__global__ void __launch_bounds__(256)
fff_output(const float* __restrict__ x, const float* __restrict__ W_in,
           const float* __restrict__ b_in, const float* __restrict__ Wt,
           const unsigned char* __restrict__ dec, float* __restrict__ out) {
    __shared__ float xs[D];
    __shared__ int   hlist[96];
    __shared__ float vals[96];

    const int tid  = threadIdx.x;
    const int wave = tid >> 5;
    const int lane = tid & 31;
    const int b    = blockIdx.x;

#pragma unroll
    for (int i = 0; i < 4; ++i) xs[tid + i * 256] = x[(size_t)b * D + tid + i * 256];

    if (tid < P_) {
        const int p = tid;
        const int p4[5] = {1, 4, 16, 64, 256};
        int cur = 0, plat = 0;
        int nodes[6];
        nodes[0] = 0;
#pragma unroll
        for (int d = 0; d < 5; ++d) {
            const int mv = (int)dec[(size_t)b * NPG + p * G_ + cur];
            const int nplat = plat + p4[d];
            cur  = nplat + FAN * (cur - plat) + mv;
            plat = nplat;
            nodes[d + 1] = cur;
        }
#pragma unroll
        for (int l = 0; l < 6; ++l)
#pragma unroll
            for (int f = 0; f < FAN; ++f)
                hlist[p * 24 + l * 4 + f] = (p * G_ + nodes[l]) * FAN + f;
    }
    __syncthreads();

    /* recompute the 96 selected activations: wave-parallel 1024-dot products */
    for (int j = wave * 12; j < wave * 12 + 12; ++j) {
        const int h = hlist[j];
        const float* wr = W_in + (size_t)h * D;
        float s = 0.0f;
#pragma unroll
        for (int i = 0; i < 8; ++i) {
            const float4 w4 = *(const float4*)(wr + lane * 4 + i * 128);
            const float4 x4 = *(const float4*)(xs + lane * 4 + i * 128);
            s += w4.x * x4.x + w4.y * x4.y + w4.z * x4.z + w4.w * x4.w;
        }
#pragma unroll
        for (int off = 16; off > 0; off >>= 1) s += __shfl_xor(s, off);
        if (lane == 0) vals[j] = gelu_exact(s + b_in[h]);
    }
    __syncthreads();

    /* out[b, :] = sum_j vals[j] * Wt[h_j, :]  (contiguous row gather, L2 hit) */
    const int d0 = tid * 4;
    float a0 = 0.0f, a1 = 0.0f, a2 = 0.0f, a3 = 0.0f;
    for (int j = 0; j < 96; ++j) {
        const float v = vals[j];
        const float4 w4 = *(const float4*)(Wt + (size_t)hlist[j] * D + d0);
        a0 += v * w4.x; a1 += v * w4.y; a2 += v * w4.z; a3 += v * w4.w;
    }
    float4 o; o.x = a0; o.y = a1; o.z = a2; o.w = a3;
    *(float4*)(out + (size_t)b * D + d0) = o;
}

/* --------------------------------- launch ---------------------------------- */
extern "C" void kernel_launch(void* const* d_in, const int* in_sizes, int n_in,
                              void* d_out, int out_size, void* d_ws, size_t ws_size,
                              hipStream_t stream) {
    (void)in_sizes; (void)n_in; (void)out_size; (void)ws_size;
    const float* oldx  = (const float*)d_in[0];   /* [2,2048,1024] */
    const float* W_in  = (const float*)d_in[1];   /* [HID, D]      */
    const float* b_in  = (const float*)d_in[2];   /* [HID]         */
    const float* W_out = (const float*)d_in[3];   /* [D, HID]      */
    float* out = (float*)d_out;

    float* Wt = (float*)d_ws;                                             /* 89.4 MB */
    unsigned char* dec = (unsigned char*)d_ws + (size_t)HID * D * sizeof(float); /* 22.4 MB */

    transpose_wout<<<dim3((HID + 31) / 32, D / 32), dim3(32, 8), 0, stream>>>(W_out, Wt);
    gemm_decide<<<dim3(HID / BN, BATCH / BM), 256, 0, stream>>>(oldx, W_in, b_in, dec);
    fff_output<<<BATCH, 256, 0, stream>>>(oldx, W_in, b_in, Wt, dec, out);
}
